// SessionGraph_71167608094726
// MI455X (gfx1250) — compile-verified, matrix operand order checked by reading, not run on hardware
//
#include <hip/hip_runtime.h>
#include <hip/hip_bf16.h>
#include <math.h>

typedef _Float16 h16_t;
typedef __attribute__((ext_vector_type(16))) _Float16 v16h;
typedef __attribute__((ext_vector_type(8)))  float    v8f;

union FragU { v16h v; unsigned int u[8]; h16_t e[16]; };

__device__ __forceinline__ int frag_k(int i, int lane) {
  return ((i >> 2) << 4) + ((lane & 16) ? 8 : 0) + ((i & 3) << 1);
}

// A/B fragment where the "tile row" dim (M for A, N for W.T-as-B) has stride ld
// and K is contiguous.  Pairs (k,k+1) are contiguous -> 32-bit loads.
__device__ __forceinline__ v16h ld_frag_rk(const h16_t* p, int ld, int lane) {
  FragU f;
  const h16_t* row = p + (lane & 15) * ld;
#pragma unroll
  for (int i = 0; i < 8; ++i)
    f.u[i] = *(const unsigned int*)(row + frag_k(i, lane));
  return f.v;
}

// B fragment from a row-major K x N matrix (stride ld between K rows).
__device__ __forceinline__ v16h ld_frag_ck(const h16_t* p, int ld, int lane) {
  FragU f;
  const h16_t* col = p + (lane & 15);
#pragma unroll
  for (int i = 0; i < 8; ++i) {
    int k = frag_k(i, lane);
    f.e[2 * i]     = col[k * ld];
    f.e[2 * i + 1] = col[(k + 1) * ld];
  }
  return f.v;
}

__device__ __forceinline__ v8f wmma16(v16h a, v16h b, v8f c) {
  return __builtin_amdgcn_wmma_f32_16x16x32_f16(false, a, false, b, (short)0, c,
                                                false, false);
}

// ---------------- prep kernels ----------------

// dst (prows x pcols, f16) = zero-padded src (rows x cols, f32)
__global__ void k_cvt_pad(const float* __restrict__ src, h16_t* __restrict__ dst,
                          int rows, int cols, int pcols, int total) {
  int i = blockIdx.x * blockDim.x + threadIdx.x;
  if (i < total) {
    int r = i / pcols, c = i - r * pcols;
    dst[i] = (r < rows && c < cols) ? (h16_t)src[r * cols + c] : (h16_t)0.f;
  }
}

// H16[row][0..127] = emb[items[row]][0..99] (f16, zero padded)
__global__ void k_gather(const int* __restrict__ items,
                         const float* __restrict__ emb,
                         h16_t* __restrict__ H16) {
  int row = blockIdx.x;
  int c = threadIdx.x;  // 128 threads
  int it = items[row];
  H16[(size_t)row * 128 + c] =
      (c < 100) ? (h16_t)emb[(size_t)it * 100 + c] : (h16_t)0.f;
}

// ---------------- generic C = A @ W.T + bias (f16 out) ----------------
// A: (Mrows x lda) f16 global, W: (ntiles*16 x ldw) f16 padded, C: (Mrows x ldc).
// grid.y selects a 64-row strip; 8 waves share 4*ntiles 16x16 tiles.
__global__ void __launch_bounds__(256) k_gemm_wT(
    const h16_t* __restrict__ A, int lda, const h16_t* __restrict__ W, int ldw,
    const float* __restrict__ bias, int nreal, h16_t* __restrict__ C, int ldc,
    int ntiles, int ksteps) {
  __shared__ h16_t sA[64 * 224];
  const int tid = threadIdx.x, lane = tid & 31, wave = tid >> 5;
  const size_t arow = (size_t)blockIdx.y * 64;
  {
    const unsigned int* src = (const unsigned int*)(A + arow * lda);
    unsigned int* dst = (unsigned int*)sA;
    int n2 = 64 * lda / 2;
    for (int i = tid; i < n2; i += 256) dst[i] = src[i];
  }
  __syncthreads();
  const int njobs = 4 * ntiles;
  for (int j = wave; j < njobs; j += 8) {
    int mt = j / ntiles, nt = j - mt * ntiles;
    v8f acc = {};
    for (int ks = 0; ks < ksteps; ++ks) {
      v16h a = ld_frag_rk(sA + (mt * 16) * lda + ks * 32, lda, lane);
      v16h b = ld_frag_rk(W + (size_t)(nt * 16) * ldw + ks * 32, ldw, lane);
      acc = wmma16(a, b, acc);
    }
    int n = lane & 15, mb = (lane & 16) ? 8 : 0;
    int col = nt * 16 + n;
    float bv = (col < nreal) ? bias[col] : 0.0f;
    h16_t* crow = C + arow * ldc;
#pragma unroll
    for (int r = 0; r < 8; ++r)
      crow[(size_t)(mt * 16 + mb + r) * ldc + col] = (h16_t)(acc[r] + bv);
  }
}

// ---------------- a_in/a_out = A[b] @ h_in/h_out + bias -> inputs ----------
__global__ void __launch_bounds__(256) k_edge(
    const float* __restrict__ Afull, const h16_t* __restrict__ HIN,
    const h16_t* __restrict__ HOUT, const float* __restrict__ biah,
    const float* __restrict__ boah, h16_t* __restrict__ INP) {
  __shared__ h16_t sA[64 * 128];
  __shared__ h16_t sHin[64 * 112];
  __shared__ h16_t sHout[64 * 112];
  const int tid = threadIdx.x, lane = tid & 31, wave = tid >> 5;
  const int b = blockIdx.x;
  const float* asrc = Afull + (size_t)b * 64 * 128;
  for (int i = tid; i < 64 * 128; i += 256) sA[i] = (h16_t)asrc[i];
  {
    const unsigned int* s1 = (const unsigned int*)(HIN + (size_t)b * 64 * 112);
    const unsigned int* s2 = (const unsigned int*)(HOUT + (size_t)b * 64 * 112);
    unsigned int* d1 = (unsigned int*)sHin;
    unsigned int* d2 = (unsigned int*)sHout;
    for (int i = tid; i < 64 * 112 / 2; i += 256) { d1[i] = s1[i]; d2[i] = s2[i]; }
  }
  __syncthreads();
  h16_t* dst = INP + (size_t)b * 64 * 224;
  for (int j = wave; j < 56; j += 8) {  // 2 gemms x 4 mtiles x 7 ntiles
    int g = j / 28, rem = j - g * 28;
    int mt = rem / 7, nt = rem - mt * 7;
    const h16_t* hb = g ? sHout : sHin;
    v8f acc = {};
#pragma unroll
    for (int ks = 0; ks < 2; ++ks) {  // K = 64
      v16h a = ld_frag_rk(sA + (mt * 16) * 128 + g * 64 + ks * 32, 128, lane);
      v16h bf = ld_frag_ck(hb + (ks * 32) * 112 + nt * 16, 112, lane);
      acc = wmma16(a, bf, acc);
    }
    int n = lane & 15, mb = (lane & 16) ? 8 : 0;
    int col = nt * 16 + n;
    if (g == 0) {
      if (col < 100) {
        float bv = biah[col];
#pragma unroll
        for (int r = 0; r < 8; ++r)
          dst[(mt * 16 + mb + r) * 224 + col] = (h16_t)(acc[r] + bv);
      }
    } else {
      float bv = (col < 100) ? boah[col] : 0.0f;
#pragma unroll
      for (int r = 0; r < 8; ++r)
        dst[(mt * 16 + mb + r) * 224 + 100 + col] = (h16_t)(acc[r] + bv);
    }
  }
  for (int i = tid; i < 64 * 12; i += 256) {  // zero cols 212..223
    int m = i / 12, c = 212 + (i - m * 12);
    dst[m * 224 + c] = (h16_t)0.f;
  }
}

// ---------------- GRU gate + LayerNorm (wave per row) ----------------
__global__ void __launch_bounds__(256) k_gru_ln(
    const h16_t* __restrict__ GI, const h16_t* __restrict__ GH,
    const h16_t* __restrict__ H, const float* __restrict__ lnw,
    const float* __restrict__ lnb, h16_t* __restrict__ HLN) {
  const int tid = threadIdx.x, lane = tid & 31, wave = tid >> 5;
  const int row = blockIdx.x * 8 + wave;
  const h16_t* gi = GI + (size_t)row * 304;
  const h16_t* gh = GH + (size_t)row * 304;
  const h16_t* h = H + (size_t)row * 128;
  float hv[4];
  float s1 = 0.f, s2 = 0.f;
#pragma unroll
  for (int it = 0; it < 4; ++it) {
    int f = lane + 32 * it;
    float v = 0.f;
    if (f < 100) {
      float ir = (float)gi[f], ii = (float)gi[100 + f], in = (float)gi[200 + f];
      float hr = (float)gh[f], hi = (float)gh[100 + f], hn = (float)gh[200 + f];
      float hh = (float)h[f];
      float r = 1.f / (1.f + expf(-(ir + hr)));
      float z = 1.f / (1.f + expf(-(ii + hi)));
      float n = tanhf(in + r * hn);
      v = (1.f - z) * hh + z * n;
      s1 += v; s2 += v * v;
    }
    hv[it] = v;
  }
#pragma unroll
  for (int off = 16; off > 0; off >>= 1) {
    s1 += __shfl_xor(s1, off, 32);
    s2 += __shfl_xor(s2, off, 32);
  }
  float mu = s1 * 0.01f;
  float var = s2 * 0.01f - mu * mu;
  float inv = rsqrtf(var + 1e-5f);
  h16_t* o = HLN + (size_t)row * 128;
#pragma unroll
  for (int it = 0; it < 4; ++it) {
    int f = lane + 32 * it;
    float v = (f < 100) ? (lnw[f] * (hv[it] - mu) * inv + lnb[f]) : 0.f;
    o[f] = (h16_t)v;
  }
}

// ---------------- attention -> sess (block per batch element) -------------
__global__ void __launch_bounds__(256) k_attn(
    const h16_t* __restrict__ HLN, const h16_t* __restrict__ Q1,
    const h16_t* __restrict__ Q2, const float* __restrict__ wz,
    const float* __restrict__ bz, const float* __restrict__ mask,
    float* __restrict__ SESS) {
  __shared__ float s_ht[100];
  __shared__ float s_q0[100];
  __shared__ float s_al[64 * 4];
  __shared__ float s_w[64 * 4];
  __shared__ float s_ms[8];
  const int tid = threadIdx.x;
  const int b = blockIdx.x;
  if (tid < 100) s_ht[tid] = (float)HLN[((size_t)b * 64 + 63) * 128 + tid];
  __syncthreads();
  if (tid < 100) {
    float s = bz[tid];
    const float* wr = wz + tid * 100;
    for (int k = 0; k < 100; ++k) s += wr[k] * s_ht[k];
    s_q0[tid] = s;
  }
  __syncthreads();
  {
    int l = tid >> 2, hh = tid & 3;
    const h16_t* q1r = Q1 + ((size_t)b * 64 + l) * 112 + hh * 25;
    float s = 0.f;
    for (int d = 0; d < 25; ++d) s += s_q0[hh * 25 + d] * (float)q1r[d];
    s_al[l * 4 + hh] = 1.f / (1.f + expf(-s));
  }
  __syncthreads();
  if (tid < 4) {
    float mx = -1e30f;
    for (int l = 0; l < 64; ++l) mx = fmaxf(mx, s_al[l * 4 + tid]);
    float sm = 0.f;
    for (int l = 0; l < 64; ++l) sm += expf(2.f * s_al[l * 4 + tid] - 2.f * mx);
    s_ms[tid] = 2.f * mx;
    s_ms[4 + tid] = sm;
  }
  __syncthreads();
  {
    int l = tid >> 2, hh = tid & 3;
    s_w[l * 4 + hh] = expf(2.f * s_al[l * 4 + hh] - s_ms[hh]) / s_ms[4 + hh];
  }
  __syncthreads();
  if (tid < 100) {
    int hh = tid / 25;
    float acc = 0.f;
    for (int l = 0; l < 64; ++l) {
      float mk = mask[b * 64 + l];
      acc += s_w[l * 4 + hh] * mk * (float)Q2[((size_t)b * 64 + l) * 112 + tid];
    }
    SESS[b * 100 + tid] = acc;
  }
}

// ---------------- norms ----------------
__global__ void k_nrm(const float* __restrict__ SESS, float* __restrict__ NRM) {
  int b = blockIdx.x * blockDim.x + threadIdx.x;
  if (b < 1024) {
    const float* s = SESS + b * 100;
    float a = 0.f;
    for (int k = 0; k < 100; ++k) a += s[k] * s[k] + 1e-6f;
    NRM[b] = sqrtf(a);
  }
}

// ---------------- cosine softmax + top5 + neighbor -> S16 ------------------
__global__ void __launch_bounds__(256) k_neighbor(
    const float* __restrict__ SESS, const float* __restrict__ NRM,
    h16_t* __restrict__ S16) {
  __shared__ float s_sb[100];
  __shared__ float s_c[1024];
  __shared__ float s_rv[256];
  __shared__ int s_ri[256];
  __shared__ float s_sel_v[5];
  __shared__ int s_sel_i[5];
  __shared__ float s_wk[5];
  const int tid = threadIdx.x;
  const int b = blockIdx.x;
  if (tid < 100) s_sb[tid] = SESS[b * 100 + tid];
  __syncthreads();
  float nb = NRM[b];
  for (int j = tid; j < 1024; j += 256) {
    const float* sj = SESS + j * 100;
    float d = 0.f;
    for (int k = 0; k < 100; ++k) d += s_sb[k] * sj[k];
    s_c[j] = d / (nb * NRM[j]);
  }
  __syncthreads();
  float lm = -1e30f;
  for (int j = tid; j < 1024; j += 256) lm = fmaxf(lm, s_c[j]);
  s_rv[tid] = lm;
  __syncthreads();
  for (int off = 128; off > 0; off >>= 1) {
    if (tid < off) s_rv[tid] = fmaxf(s_rv[tid], s_rv[tid + off]);
    __syncthreads();
  }
  float M = s_rv[0];
  __syncthreads();
  float ls = 0.f;
  for (int j = tid; j < 1024; j += 256) {
    float e = expf(s_c[j] - M);
    s_c[j] = e;
    ls += e;
  }
  s_rv[tid] = ls;
  __syncthreads();
  for (int off = 128; off > 0; off >>= 1) {
    if (tid < off) s_rv[tid] += s_rv[tid + off];
    __syncthreads();
  }
  float S = s_rv[0];
  __syncthreads();
  for (int j = tid; j < 1024; j += 256) s_c[j] /= S;
  __syncthreads();
  for (int t = 0; t < 5; ++t) {
    float bv = -1e30f; int bi = 1 << 30;
    for (int j = tid; j < 1024; j += 256) {
      float v = s_c[j];
      if (v > bv || (v == bv && j < bi)) { bv = v; bi = j; }
    }
    s_rv[tid] = bv; s_ri[tid] = bi;
    __syncthreads();
    for (int off = 128; off > 0; off >>= 1) {
      if (tid < off) {
        float v2 = s_rv[tid + off]; int i2 = s_ri[tid + off];
        if (v2 > s_rv[tid] || (v2 == s_rv[tid] && i2 < s_ri[tid])) {
          s_rv[tid] = v2; s_ri[tid] = i2;
        }
      }
      __syncthreads();
    }
    if (tid == 0) { s_sel_v[t] = s_rv[0]; s_sel_i[t] = s_ri[0]; s_c[s_ri[0]] = -1e30f; }
    __syncthreads();
  }
  if (tid == 0) {
    float m0 = s_sel_v[0];
    float ssum = 0.f;
    for (int t = 0; t < 5; ++t) ssum += expf(s_sel_v[t] - m0);
    for (int t = 0; t < 5; ++t) s_wk[t] = expf(s_sel_v[t] - m0) / ssum;
  }
  __syncthreads();
  if (tid < 128) {
    float acc = 0.f;
    if (tid < 100) {
      acc = s_sb[tid];
      for (int t = 0; t < 5; ++t) acc += s_wk[t] * SESS[s_sel_i[t] * 100 + tid];
    }
    S16[b * 128 + tid] = (h16_t)acc;
  }
}

// ---------------- scores = 20 * S @ emb.T  (M=1024, N=100000, K=128) -------
__global__ void __launch_bounds__(256) k_scores(const h16_t* __restrict__ S16,
                                                const h16_t* __restrict__ E16,
                                                float* __restrict__ out) {
  __shared__ h16_t sA[64 * 128];
  const int tid = threadIdx.x, lane = tid & 31, wave = tid >> 5;
  const int m0 = blockIdx.y * 64;
  {
    const unsigned int* src = (const unsigned int*)(S16 + (size_t)m0 * 128);
    unsigned int* dst = (unsigned int*)sA;
    for (int i = tid; i < 64 * 128 / 2; i += 256) dst[i] = src[i];
  }
  __syncthreads();
  int ntb = 6250 - blockIdx.x * 16;
  if (ntb > 16) ntb = 16;
  for (int j = wave; j < ntb; j += 8) {
    int n0 = (blockIdx.x * 16 + j) * 16;
    v8f acc[4] = {};
    for (int ks = 0; ks < 4; ++ks) {
      v16h bf = ld_frag_rk(E16 + (size_t)n0 * 128 + ks * 32, 128, lane);
#pragma unroll
      for (int mt = 0; mt < 4; ++mt) {
        v16h af = ld_frag_rk(sA + (mt * 16) * 128 + ks * 32, 128, lane);
        acc[mt] = wmma16(af, bf, acc[mt]);
      }
    }
    int n = lane & 15, mb = (lane & 16) ? 8 : 0;
    int col = n0 + n;
#pragma unroll
    for (int mt = 0; mt < 4; ++mt)
#pragma unroll
      for (int r = 0; r < 8; ++r)
        out[(size_t)(m0 + mt * 16 + mb + r) * 100000 + col] = 20.0f * acc[mt][r];
  }
}

// ---------------- host orchestration ----------------
extern "C" void kernel_launch(void* const* d_in, const int* in_sizes, int n_in,
                              void* d_out, int out_size, void* d_ws,
                              size_t ws_size, hipStream_t stream) {
  (void)in_sizes; (void)n_in; (void)out_size; (void)ws_size;
  const int* items   = (const int*)d_in[0];
  const float* Amat  = (const float*)d_in[1];
  const float* mask  = (const float*)d_in[2];
  const float* emb   = (const float*)d_in[3];
  const float* w_in  = (const float*)d_in[4];
  const float* b_in  = (const float*)d_in[5];
  const float* w_out = (const float*)d_in[6];
  const float* b_out = (const float*)d_in[7];
  const float* b_iah = (const float*)d_in[8];
  const float* b_oah = (const float*)d_in[9];
  const float* w_ih  = (const float*)d_in[10];
  const float* b_ih  = (const float*)d_in[11];
  const float* w_hh  = (const float*)d_in[12];
  const float* b_hh  = (const float*)d_in[13];
  const float* ln_w  = (const float*)d_in[14];
  const float* ln_b  = (const float*)d_in[15];
  const float* wz    = (const float*)d_in[16];
  const float* bz    = (const float*)d_in[17];
  const float* w1    = (const float*)d_in[18];
  const float* b1    = (const float*)d_in[19];
  const float* w2    = (const float*)d_in[20];
  const float* b2    = (const float*)d_in[21];
  float* out = (float*)d_out;

  char* ws = (char*)d_ws;
  size_t off = 0;
  auto alloc = [&](size_t bytes) -> char* {
    char* p = ws + off;
    off = (off + bytes + 255) & ~(size_t)255;
    return p;
  };
  h16_t* W_IN  = (h16_t*)alloc((size_t)112 * 128 * 2);
  h16_t* W_OUT = (h16_t*)alloc((size_t)112 * 128 * 2);
  h16_t* W_IH  = (h16_t*)alloc((size_t)304 * 224 * 2);
  h16_t* W_HH  = (h16_t*)alloc((size_t)304 * 128 * 2);
  h16_t* W_1   = (h16_t*)alloc((size_t)112 * 128 * 2);
  h16_t* W_2   = (h16_t*)alloc((size_t)112 * 128 * 2);
  h16_t* EMB16 = (h16_t*)alloc((size_t)100000 * 128 * 2);
  h16_t* H16   = (h16_t*)alloc((size_t)65536 * 128 * 2);
  h16_t* HIN   = (h16_t*)alloc((size_t)65536 * 112 * 2);
  h16_t* HOUT  = (h16_t*)alloc((size_t)65536 * 112 * 2);
  h16_t* INP   = (h16_t*)alloc((size_t)65536 * 224 * 2);
  h16_t* GI    = (h16_t*)alloc((size_t)65536 * 304 * 2);
  h16_t* GH    = (h16_t*)alloc((size_t)65536 * 304 * 2);
  h16_t* HLN   = (h16_t*)alloc((size_t)65536 * 128 * 2);
  h16_t* Q1    = (h16_t*)alloc((size_t)65536 * 112 * 2);
  h16_t* Q2    = (h16_t*)alloc((size_t)65536 * 112 * 2);
  float* SESS  = (float*)alloc((size_t)1024 * 100 * 4);
  float* NRM   = (float*)alloc((size_t)1024 * 4);
  h16_t* S16   = (h16_t*)alloc((size_t)1024 * 128 * 2);

  auto pad = [&](const float* s, h16_t* d, int r, int c, int pr, int pc) {
    int total = pr * pc;
    k_cvt_pad<<<(total + 255) / 256, 256, 0, stream>>>(s, d, r, c, pc, total);
  };
  pad(w_in, W_IN, 100, 100, 112, 128);
  pad(w_out, W_OUT, 100, 100, 112, 128);
  pad(w_ih, W_IH, 300, 200, 304, 224);
  pad(w_hh, W_HH, 300, 100, 304, 128);
  pad(w1, W_1, 100, 100, 112, 128);
  pad(w2, W_2, 100, 100, 112, 128);
  pad(emb, EMB16, 100000, 100, 100000, 128);

  k_gather<<<65536, 128, 0, stream>>>(items, emb, H16);

  dim3 g1(1, 1024);
  k_gemm_wT<<<g1, 256, 0, stream>>>(H16, 128, W_IN, 128, b_in, 100, HIN, 112, 7, 4);
  k_gemm_wT<<<g1, 256, 0, stream>>>(H16, 128, W_OUT, 128, b_out, 100, HOUT, 112, 7, 4);
  k_edge<<<1024, 256, 0, stream>>>(Amat, HIN, HOUT, b_iah, b_oah, INP);
  k_gemm_wT<<<g1, 256, 0, stream>>>(INP, 224, W_IH, 224, b_ih, 300, GI, 304, 19, 7);
  k_gemm_wT<<<g1, 256, 0, stream>>>(H16, 128, W_HH, 128, b_hh, 300, GH, 304, 19, 4);
  k_gru_ln<<<8192, 256, 0, stream>>>(GI, GH, H16, ln_w, ln_b, HLN);
  k_gemm_wT<<<g1, 256, 0, stream>>>(HLN, 128, W_1, 128, b1, 100, Q1, 112, 7, 4);
  k_gemm_wT<<<g1, 256, 0, stream>>>(HLN, 128, W_2, 128, b2, 100, Q2, 112, 7, 4);
  k_attn<<<1024, 256, 0, stream>>>(HLN, Q1, Q2, wz, bz, mask, SESS);
  k_nrm<<<4, 256, 0, stream>>>(SESS, NRM);
  k_neighbor<<<1024, 256, 0, stream>>>(SESS, NRM, S16);
  dim3 g3(391, 16);
  k_scores<<<g3, 256, 0, stream>>>(S16, EMB16, out);
}
